// EdgeUpdater_30305289240588
// MI455X (gfx1250) — matrix operand order repaired and checked by reading, not compiled, not run on hardware
//
#include <hip/hip_runtime.h>
#include <hip/hip_bf16.h>
#include <stdint.h>

typedef __attribute__((ext_vector_type(16))) __bf16 v16bf;
typedef __attribute__((ext_vector_type(8)))  __bf16 v8bf;
typedef __attribute__((ext_vector_type(8)))  float  v8f;
typedef __attribute__((ext_vector_type(4)))  float  v4f;

#define HID 128

// Fast transcendental helpers for gfx1250 -----------------------------------
__device__ __forceinline__ float fast_tanh(float x) {
#if __has_builtin(__builtin_amdgcn_tanhf)
    return __builtin_amdgcn_tanhf(x);
#elif defined(__AMDGCN__)
    float r;
    asm volatile("v_tanh_f32 %0, %1" : "=v"(r) : "v"(x));
    return r;
#else
    return tanhf(x);
#endif
}

__device__ __forceinline__ float fast_silu(float v) {
    // v * sigmoid(v) with hardware rcp (1 trans op) instead of IEEE divide
    return v * __builtin_amdgcn_rcpf(1.f + __expf(-v));
}

// ---------------------------------------------------------------------------
// Pack W[k][n] (f32, row-major, n-stride 128) into bf16 B-fragment layout:
// tile (kt,nt) covers K=[kt*32,kt*32+32), N=[nt*16,nt*16+16).
// Per lane: col n = nt*16 + (lane&15); elems i=0..15 -> K = kt*32 + (lane>>4)*8 + (i&7) + (i>>3)*16.
// Linear: out[((kt*8+nt)*32 + lane)*16 + i]
// ---------------------------------------------------------------------------
__global__ void pack_w_kernel(const float* __restrict__ W, unsigned short* __restrict__ outp,
                              int ktiles) {
    int idx = blockIdx.x * blockDim.x + threadIdx.x;
    int total = ktiles * 8 * 512;
    if (idx >= total) return;
    int t    = idx >> 9;          // tile index
    int lane = (idx >> 4) & 31;
    int i    = idx & 15;
    int kt = t >> 3, nt = t & 7;
    int k = kt * 32 + ((lane >> 4) << 3) + (i & 7) + ((i >> 3) << 4);
    int n = nt * 16 + (lane & 15);
    __bf16 v = (__bf16)W[k * HID + n];
    ((__bf16*)outp)[idx] = v;
}

// ---------------------------------------------------------------------------
// Main fused kernel: 128 threads = 4 waves, 32 edges per wave (2 M-tiles).
// ---------------------------------------------------------------------------
__launch_bounds__(128, 1)
__global__ void edge_mlp_kernel(const float* __restrict__ x,
                                const int64_t* __restrict__ ei,
                                const float* __restrict__ edgef,
                                const float* __restrict__ b1, const float* __restrict__ g1,
                                const float* __restrict__ be1,
                                const float* __restrict__ b2, const float* __restrict__ g2,
                                const float* __restrict__ be2,
                                const unsigned short* __restrict__ w1p,
                                const unsigned short* __restrict__ w2p,
                                float* __restrict__ out, int E) {
    __shared__ __bf16 sh[4 * 32 * HID];   // 32 KB: per-wave 32x128 bf16 staging

    const __bf16* w1f = (const __bf16*)w1p;
    const __bf16* w2f = (const __bf16*)w2p;

    const int lane  = threadIdx.x & 31;
    const int wave  = threadIdx.x >> 5;
    const int row   = lane & 15;      // A-row / B-col / C-col selector
    const int hi    = lane >> 4;      // lane half
    const int kb8   = hi * 8;         // A-fragment K sub-base

    const int eWave = blockIdx.x * 128 + wave * 32;
    __bf16* my = sh + wave * 32 * HID;

    // Per-lane source pointers for the two M-tiles (clamped for tail)
    const float* pS[2]; const float* pD[2]; const float* pE[2];
#pragma unroll
    for (int mt = 0; mt < 2; ++mt) {
        int e  = eWave + mt * 16 + row;
        int ec = e < E ? e : (E - 1);
        int64_t s = ei[ec];
        int64_t d = ei[(int64_t)E + ec];
        pS[mt] = x + (size_t)s * HID;
        pD[mt] = x + (size_t)d * HID;
        pE[mt] = edgef + (size_t)ec * HID;
    }

    v8f acc[2][8];
#pragma unroll
    for (int mt = 0; mt < 2; ++mt)
#pragma unroll
        for (int nt = 0; nt < 8; ++nt) acc[mt][nt] = (v8f){};

    // ---------------- GEMM1: [16x384] @ [384x128] over 12 K-tiles ----------
    for (int kt = 0; kt < 12; ++kt) {
        int koff = (kt & 3) * 32;
        v16bf a[2];
#pragma unroll
        for (int mt = 0; mt < 2; ++mt) {
            const float* p = (kt < 4) ? pS[mt] : ((kt < 8) ? pD[mt] : pE[mt]);
            const float* q = p + koff + kb8;
            v4f f0 = *(const v4f*)(q);
            v4f f1 = *(const v4f*)(q + 4);
            v4f f2 = *(const v4f*)(q + 16);
            v4f f3 = *(const v4f*)(q + 20);
            v16bf af;
#pragma unroll
            for (int j = 0; j < 4; ++j) {
                af[j]      = (__bf16)f0[j];
                af[4 + j]  = (__bf16)f1[j];
                af[8 + j]  = (__bf16)f2[j];
                af[12 + j] = (__bf16)f3[j];
            }
            a[mt] = af;
        }
        const __bf16* wt = w1f + (size_t)kt * 8 * 512 + lane * 16;
#pragma unroll
        for (int nt = 0; nt < 8; ++nt) {
            v16bf b = *(const v16bf*)(wt + nt * 512);
#pragma unroll
            for (int mt = 0; mt < 2; ++mt)
                acc[mt][nt] = __builtin_amdgcn_wmma_f32_16x16x32_bf16(
                    false, a[mt], false, b, (short)0, acc[mt][nt], false, false);
        }
    }

    // ---------------- bias + LN1 + SiLU -> LDS (bf16, row-major) -----------
    float gv[8], bv[8], biasv[8];
#pragma unroll
    for (int nt = 0; nt < 8; ++nt) {
        int c = nt * 16 + row;
        gv[nt] = g1[c]; bv[nt] = be1[c]; biasv[nt] = b1[c];
    }
#pragma unroll
    for (int mt = 0; mt < 2; ++mt) {
#pragma unroll
        for (int r = 0; r < 8; ++r) {
            float s = 0.f, q2 = 0.f;
#pragma unroll
            for (int nt = 0; nt < 8; ++nt) {
                float v = acc[mt][nt][r] + biasv[nt];
                acc[mt][nt][r] = v;
                s += v; q2 += v * v;
            }
            s  += __shfl_xor(s, 1);  s  += __shfl_xor(s, 2);
            s  += __shfl_xor(s, 4);  s  += __shfl_xor(s, 8);
            q2 += __shfl_xor(q2, 1); q2 += __shfl_xor(q2, 2);
            q2 += __shfl_xor(q2, 4); q2 += __shfl_xor(q2, 8);
            float mean = s * (1.f / 128.f);
            float var  = q2 * (1.f / 128.f) - mean * mean;
            float inv  = rsqrtf(var + 1e-5f);
            int rowLocal = mt * 16 + r + hi * 8;
#pragma unroll
            for (int nt = 0; nt < 8; ++nt) {
                float v = (acc[mt][nt][r] - mean) * inv * gv[nt] + bv[nt];
                v = fast_silu(v);
                my[rowLocal * HID + nt * 16 + row] = (__bf16)v;
            }
        }
    }

    // ---------------- GEMM2: [16x128] @ [128x128], A from LDS --------------
#pragma unroll
    for (int mt = 0; mt < 2; ++mt)
#pragma unroll
        for (int nt = 0; nt < 8; ++nt) acc[mt][nt] = (v8f){};

    for (int kt = 0; kt < 4; ++kt) {
        int koff = kt * 32;
        v16bf a[2];
#pragma unroll
        for (int mt = 0; mt < 2; ++mt) {
            const __bf16* p = my + (mt * 16 + row) * HID + koff + kb8;
            v8bf lo  = *(const v8bf*)(p);
            v8bf hi2 = *(const v8bf*)(p + 16);
            v16bf af;
#pragma unroll
            for (int j = 0; j < 8; ++j) { af[j] = lo[j]; af[8 + j] = hi2[j]; }
            a[mt] = af;
        }
        const __bf16* wt = w2f + (size_t)kt * 8 * 512 + lane * 16;
#pragma unroll
        for (int nt = 0; nt < 8; ++nt) {
            v16bf b = *(const v16bf*)(wt + nt * 512);
#pragma unroll
            for (int mt = 0; mt < 2; ++mt)
                acc[mt][nt] = __builtin_amdgcn_wmma_f32_16x16x32_bf16(
                    false, a[mt], false, b, (short)0, acc[mt][nt], false, false);
        }
    }

    // ---------------- bias + LN2 + tanh + residual -> out ------------------
#pragma unroll
    for (int nt = 0; nt < 8; ++nt) {
        int c = nt * 16 + row;
        gv[nt] = g2[c]; bv[nt] = be2[c]; biasv[nt] = b2[c];
    }
#pragma unroll
    for (int mt = 0; mt < 2; ++mt) {
#pragma unroll
        for (int r = 0; r < 8; ++r) {
            float s = 0.f, q2 = 0.f;
#pragma unroll
            for (int nt = 0; nt < 8; ++nt) {
                float v = acc[mt][nt][r] + biasv[nt];
                acc[mt][nt][r] = v;
                s += v; q2 += v * v;
            }
            s  += __shfl_xor(s, 1);  s  += __shfl_xor(s, 2);
            s  += __shfl_xor(s, 4);  s  += __shfl_xor(s, 8);
            q2 += __shfl_xor(q2, 1); q2 += __shfl_xor(q2, 2);
            q2 += __shfl_xor(q2, 4); q2 += __shfl_xor(q2, 8);
            float mean = s * (1.f / 128.f);
            float var  = q2 * (1.f / 128.f) - mean * mean;
            float inv  = rsqrtf(var + 1e-5f);
            int e = eWave + mt * 16 + r + hi * 8;   // C-layout row -> edge id
            if (e < E) {
#pragma unroll
                for (int nt = 0; nt < 8; ++nt) {
                    float v = (acc[mt][nt][r] - mean) * inv * gv[nt] + bv[nt];
                    v = fast_tanh(v);
                    size_t off = (size_t)e * HID + nt * 16 + row;
                    out[off] = edgef[off] + v;
                }
            }
        }
    }
}

// ---------------------------------------------------------------------------
extern "C" void kernel_launch(void* const* d_in, const int* in_sizes, int n_in,
                              void* d_out, int out_size, void* d_ws, size_t ws_size,
                              hipStream_t stream) {
    (void)n_in; (void)out_size; (void)ws_size;
    const float*   x    = (const float*)d_in[0];
    const int64_t* ei   = (const int64_t*)d_in[1];
    const float*   edge = (const float*)d_in[2];
    const float*   W1   = (const float*)d_in[3];
    const float*   b1   = (const float*)d_in[4];
    const float*   g1   = (const float*)d_in[5];
    const float*   be1  = (const float*)d_in[6];
    const float*   W2   = (const float*)d_in[7];
    const float*   b2   = (const float*)d_in[8];
    const float*   g2   = (const float*)d_in[9];
    const float*   be2  = (const float*)d_in[10];
    float* out = (float*)d_out;

    int E = in_sizes[2] / HID;   // edge features: E x 128

    unsigned short* w1f = (unsigned short*)d_ws;             // 12*8*512 bf16 = 96 KB
    unsigned short* w2f = w1f + 12 * 8 * 512;                // 4*8*512 bf16  = 32 KB

    {
        int total = 12 * 8 * 512;
        pack_w_kernel<<<(total + 255) / 256, 256, 0, stream>>>(W1, w1f, 12);
    }
    {
        int total = 4 * 8 * 512;
        pack_w_kernel<<<(total + 255) / 256, 256, 0, stream>>>(W2, w2f, 4);
    }

    int nWG = (E + 127) / 128;   // 128 edges per workgroup (4 waves x 32)
    edge_mlp_kernel<<<nWG, 128, 0, stream>>>(x, ei, edge, b1, g1, be1,
                                             b2, g2, be2, w1f, w2f, out, E);
}
